// MoDLayer_7593502179631
// MI455X (gfx1250) — matrix operand order repaired and checked by reading, not compiled, not run on hardware
//
#include <hip/hip_runtime.h>
#include <hip/hip_bf16.h>

// ---------------- problem constants ----------------
#define B_    4
#define S_    4096
#define D_    2048
#define H_    16
#define KVH_  4
#define HD_   128
#define FF_   5632
#define KSEL  512            // gamma * S
#define MTOT  (B_ * KSEL)    // 2048 selected rows total

typedef __attribute__((ext_vector_type(16))) __bf16 v16bf;
typedef __attribute__((ext_vector_type(8)))  float  v8f;

// fp32 -> bf16 round-to-nearest-even
__device__ __forceinline__ unsigned short f2bf(float f) {
    unsigned int u = __float_as_uint(f);
    unsigned int r = u + 0x7FFFu + ((u >> 16) & 1u);
    return (unsigned short)(r >> 16);
}

union FragBf { uint4 q[2]; v16bf v; };

// A-matrix fragment (16x32 bf16): lane<16 -> K {0..7,16..23}; lane>=16 -> K {8..15,24..31}
__device__ __forceinline__ v16bf frag_ld_a(const unsigned short* row, int lane) {
    FragBf f; int kh = (lane >> 4) * 8;
    f.q[0] = *reinterpret_cast<const uint4*>(row + kh);
    f.q[1] = *reinterpret_cast<const uint4*>(row + 16 + kh);
    return f.v;
}
// B-matrix fragment (32x16 bf16): lane<16 -> K {0..7,16..23}; lane>=16 -> K {8..15,24..31}
__device__ __forceinline__ v16bf frag_ld_b(const unsigned short* row, int lane) {
    FragBf f; int ko = (lane >> 4) * 8;
    f.q[0] = *reinterpret_cast<const uint4*>(row + ko);
    f.q[1] = *reinterpret_cast<const uint4*>(row + 16 + ko);
    return f.v;
}

__device__ __forceinline__ v8f wmma_bf16(v16bf a, v16bf b, v8f c) {
    return __builtin_amdgcn_wmma_f32_16x16x32_bf16(false, a, false, b, (short)0, c, false, false);
}

// CDNA5 async global->LDS copy (16B per lane), tracked by ASYNCcnt.
__device__ __forceinline__ void async_g2l_b128(unsigned lds_byte_addr, const void* gptr) {
    asm volatile("global_load_async_to_lds_b128 %0, %1, off"
                 :: "v"(lds_byte_addr), "v"(gptr) : "memory");
}
__device__ __forceinline__ void wait_asynccnt0() {
    asm volatile("s_wait_asynccnt 0" ::: "memory");
}

// ---------------- router + top-k (one block per batch) ----------------
__global__ __launch_bounds__(512) void router_topk_kernel(
    const float* __restrict__ hs, const float* __restrict__ wr,
    const float* __restrict__ br, int* __restrict__ selIdx, float* __restrict__ rwSel)
{
    __shared__ float wr_s[D_];
    __shared__ float rw_s[S_];
    __shared__ float key_s[S_];
    __shared__ int   idx_s[S_];
    __shared__ int   sel_s[KSEL];
    const int b = blockIdx.x, tid = threadIdx.x;
    for (int c = tid; c < D_; c += 512) wr_s[c] = wr[c];
    __syncthreads();
    const int lane = tid & 31, wave = tid >> 5;      // 16 waves
    const float* base = hs + (size_t)b * S_ * D_;
    const float bias = br[0];
    for (int s = wave; s < S_; s += 16) {
        const float* rowp = base + (size_t)s * D_;
        float acc = 0.f;
        for (int c = lane * 4; c < D_; c += 128) {
            float4 x = *reinterpret_cast<const float4*>(rowp + c);
            acc += x.x * wr_s[c] + x.y * wr_s[c + 1] + x.z * wr_s[c + 2] + x.w * wr_s[c + 3];
        }
        for (int off = 16; off > 0; off >>= 1) acc += __shfl_xor(acc, off, 32);
        if (lane == 0) rw_s[s] = acc + bias;
    }
    __syncthreads();
    for (int s = tid; s < S_; s += 512) { key_s[s] = rw_s[s]; idx_s[s] = s; }
    __syncthreads();
    // bitonic sort descending by key
    for (int k2 = 2; k2 <= S_; k2 <<= 1) {
        for (int j = k2 >> 1; j > 0; j >>= 1) {
            for (int t = tid; t < S_ / 2; t += 512) {
                int i = ((t / j) * (j << 1)) + (t & (j - 1));
                int l = i + j;
                bool up = ((i & k2) == 0);
                float ki = key_s[i], kl = key_s[l];
                bool sw = up ? (ki < kl) : (ki > kl);
                if (sw) {
                    key_s[i] = kl; key_s[l] = ki;
                    int tmp = idx_s[i]; idx_s[i] = idx_s[l]; idx_s[l] = tmp;
                }
            }
            __syncthreads();
        }
    }
    if (tid < KSEL) sel_s[tid] = idx_s[tid];
    __syncthreads();
    // sort the 512 selected indices ascending
    for (int k2 = 2; k2 <= KSEL; k2 <<= 1) {
        for (int j = k2 >> 1; j > 0; j >>= 1) {
            if (tid < KSEL / 2) {
                int i = ((tid / j) * (j << 1)) + (tid & (j - 1));
                int l = i + j;
                bool up = ((i & k2) == 0);
                int a = sel_s[i], c2 = sel_s[l];
                bool sw = up ? (a > c2) : (a < c2);
                if (sw) { sel_s[i] = c2; sel_s[l] = a; }
            }
            __syncthreads();
        }
    }
    if (tid < KSEL) {
        int si = sel_s[tid];
        selIdx[b * KSEL + tid] = si;
        rwSel[b * KSEL + tid]  = rw_s[si];
    }
}

// ---------------- weight fp32 KxN -> bf16 NxK (transposed) ----------------
__global__ __launch_bounds__(256) void w_transpose_bf16(
    const float* __restrict__ W, unsigned short* __restrict__ Wt, int K, int N)
{
    __shared__ float tile[32][33];
    const int n0 = blockIdx.x * 32, k0 = blockIdx.y * 32;
    const int tx = threadIdx.x, ty = threadIdx.y;
    for (int i = 0; i < 4; ++i) {
        int kk = ty + i * 8;
        tile[kk][tx] = W[(size_t)(k0 + kk) * N + n0 + tx];
    }
    __syncthreads();
    for (int i = 0; i < 4; ++i) {
        int nn = ty + i * 8;
        Wt[(size_t)(n0 + nn) * K + k0 + tx] = f2bf(tile[tx][nn]);
    }
}

// ---------------- gather selected rows + RMSNorm1 -> bf16 ----------------
__global__ __launch_bounds__(256) void gather_norm_kernel(
    const float* __restrict__ hs, const int* __restrict__ selIdx,
    const float* __restrict__ ln, float* __restrict__ xSel, unsigned short* __restrict__ hBf)
{
    __shared__ float red[256];
    const int tk = blockIdx.x;          // 0..MTOT-1
    const int b  = tk >> 9;             // /KSEL
    const int tid = threadIdx.x;
    const int src = selIdx[tk];
    const float* rowp = hs + ((size_t)b * S_ + src) * D_;
    float xv[8]; float acc = 0.f;
    for (int i = 0; i < 8; ++i) {
        int c = tid + i * 256;
        float x = rowp[c];
        xv[i] = x; acc += x * x;
        xSel[(size_t)tk * D_ + c] = x;
    }
    red[tid] = acc; __syncthreads();
    for (int s = 128; s > 0; s >>= 1) { if (tid < s) red[tid] += red[tid + s]; __syncthreads(); }
    float rinv = rsqrtf(red[0] / (float)D_ + 1e-6f);
    for (int i = 0; i < 8; ++i) {
        int c = tid + i * 256;
        hBf[(size_t)tk * D_ + c] = f2bf(xv[i] * rinv * ln[c]);
    }
}

// ---------------- residual add + RMSNorm2 -> bf16 ----------------
__global__ __launch_bounds__(256) void resid_norm2_kernel(
    const float* __restrict__ xSel, const float* __restrict__ attnProj,
    const float* __restrict__ ln, float* __restrict__ x2, unsigned short* __restrict__ h2Bf)
{
    __shared__ float red[256];
    const int tk = blockIdx.x, tid = threadIdx.x;
    float xv[8]; float acc = 0.f;
    for (int i = 0; i < 8; ++i) {
        int c = tid + i * 256;
        float x = xSel[(size_t)tk * D_ + c] + attnProj[(size_t)tk * D_ + c];
        xv[i] = x; acc += x * x;
        x2[(size_t)tk * D_ + c] = x;
    }
    red[tid] = acc; __syncthreads();
    for (int s = 128; s > 0; s >>= 1) { if (tid < s) red[tid] += red[tid + s]; __syncthreads(); }
    float rinv = rsqrtf(red[0] / (float)D_ + 1e-6f);
    for (int i = 0; i < 8; ++i) {
        int c = tid + i * 256;
        h2Bf[(size_t)tk * D_ + c] = f2bf(xv[i] * rinv * ln[c]);
    }
}

// ---------------- bf16 WMMA GEMM: C[MxN] = A[MxK] * Bt[NxK]^T + bias ----------------
// 256 threads = 8 waves; block tile 128x128; wave tile 32x64; K step 32.
// Double-buffered LDS tiles filled by async global->LDS copies (ASYNCcnt),
// overlapped with WMMA compute on the other buffer.
__global__ __launch_bounds__(256) void gemm_bf16_kernel(
    const unsigned short* __restrict__ A, const unsigned short* __restrict__ Bt,
    const float* __restrict__ bias, float* __restrict__ C, int M, int N, int K)
{
    constexpr int LDK = 40;           // 32 + 8 pad (bf16 elems)
    constexpr int BUFE = 128 * LDK;   // elements per buffer
    __shared__ unsigned short As[2 * BUFE];
    __shared__ unsigned short Bs[2 * BUFE];
    const int tid = threadIdx.x;
    const int lane = tid & 31, wave = tid >> 5;
    const int wm = wave >> 1, wn = wave & 1;
    const int m0 = blockIdx.y * 128, n0 = blockIdx.x * 128;
    const int lrow = tid >> 2, lch = (tid & 3) * 8;

    // per-thread LDS staging addresses (byte addresses; generic ptr low 32 bits = LDS offset)
    const unsigned ldsA0 = (unsigned)(uintptr_t)&As[lrow * LDK + lch];
    const unsigned ldsA1 = (unsigned)(uintptr_t)&As[(lrow + 64) * LDK + lch];
    const unsigned ldsB0 = (unsigned)(uintptr_t)&Bs[lrow * LDK + lch];
    const unsigned ldsB1 = (unsigned)(uintptr_t)&Bs[(lrow + 64) * LDK + lch];
    const unsigned bufBytes = (unsigned)(BUFE * sizeof(unsigned short));

    auto stage = [&](int kb, int buf) {
        unsigned boff = buf ? bufBytes : 0u;
        async_g2l_b128(ldsA0 + boff, &A[(size_t)(m0 + lrow) * K + kb + lch]);
        async_g2l_b128(ldsA1 + boff, &A[(size_t)(m0 + lrow + 64) * K + kb + lch]);
        async_g2l_b128(ldsB0 + boff, &Bt[(size_t)(n0 + lrow) * K + kb + lch]);
        async_g2l_b128(ldsB1 + boff, &Bt[(size_t)(n0 + lrow + 64) * K + kb + lch]);
    };

    v8f acc[2][4] = {};
    stage(0, 0);
    wait_asynccnt0();
    __syncthreads();

    int buf = 0;
    for (int kb = 0; kb < K; kb += 32) {
        const bool more = (kb + 32 < K);
        if (more) {
            stage(kb + 32, buf ^ 1);            // async copy of next tile (no wait yet)
            if (kb + 64 < K) {                  // keep HW prefetcher two tiles ahead
                __builtin_prefetch(&A[(size_t)(m0 + lrow) * K + kb + 64 + lch], 0, 1);
                __builtin_prefetch(&Bt[(size_t)(n0 + lrow) * K + kb + 64 + lch], 0, 1);
            }
        }
        const unsigned short* Ab = &As[buf * BUFE];
        const unsigned short* Bb = &Bs[buf * BUFE];
        v16bf af[2], bfr[4];
        for (int mt = 0; mt < 2; ++mt)
            af[mt] = frag_ld_a(&Ab[(wm * 32 + mt * 16 + (lane & 15)) * LDK], lane);
        for (int nt = 0; nt < 4; ++nt)
            bfr[nt] = frag_ld_b(&Bb[(wn * 64 + nt * 16 + (lane & 15)) * LDK], lane);
        for (int mt = 0; mt < 2; ++mt)
            for (int nt = 0; nt < 4; ++nt)
                acc[mt][nt] = wmma_bf16(af[mt], bfr[nt], acc[mt][nt]);
        if (more) wait_asynccnt0();             // next tile fully in LDS
        __syncthreads();                        // all waves done reading old buffer
        buf ^= 1;
    }

    const int nlo = lane & 15, hi = lane >> 4;
    for (int nt = 0; nt < 4; ++nt) {
        int col = n0 + wn * 64 + nt * 16 + nlo;
        float bv = bias ? bias[col] : 0.0f;
        for (int mt = 0; mt < 2; ++mt)
            for (int r = 0; r < 8; ++r) {
                int row = m0 + wm * 32 + mt * 16 + r + 8 * hi;
                C[(size_t)row * N + col] = acc[mt][nt][r] + bv;
            }
    }
}

// ---------------- RoPE on Q (adds 1/sqrt(HD) scale), write bf16 (b,h,t,d) ----------------
__global__ __launch_bounds__(128) void rope_q_kernel(
    const float* __restrict__ qRaw, const int* __restrict__ selIdx, unsigned short* __restrict__ qBf)
{
    const int h = blockIdx.x % H_;
    const int t = (blockIdx.x / H_) % KSEL;
    const int b = blockIdx.x / (H_ * KSEL);
    const int d = threadIdx.x;
    const float pos = (float)selIdx[b * KSEL + t];
    const int fi = d & 63;
    const float freq = __powf(10000.0f, -((float)(2 * fi) / (float)HD_));
    const float ang = pos * freq;
    const float c = __cosf(ang), sn = __sinf(ang);
    const float* qr = qRaw + ((size_t)(b * KSEL + t)) * D_ + h * HD_;
    float x  = qr[d];
    float xo = (d < 64) ? -qr[d + 64] : qr[d - 64];
    float val = (x * c + xo * sn) * 0.08838834764831845f;  // 1/sqrt(128)
    qBf[(((size_t)(b * H_ + h)) * KSEL + t) * HD_ + d] = f2bf(val);
}

// ---------------- RoPE on K + transpose V -> (b,kvh,d,t) bf16 ----------------
__global__ __launch_bounds__(128) void rope_kv_kernel(
    const float* __restrict__ kRaw, const float* __restrict__ vRaw,
    const int* __restrict__ selIdx, unsigned short* __restrict__ kBf, unsigned short* __restrict__ vTBf)
{
    const int kh = blockIdx.x % KVH_;
    const int t  = (blockIdx.x / KVH_) % KSEL;
    const int b  = blockIdx.x / (KVH_ * KSEL);
    const int d  = threadIdx.x;
    const float pos = (float)selIdx[b * KSEL + t];
    const int fi = d & 63;
    const float freq = __powf(10000.0f, -((float)(2 * fi) / (float)HD_));
    const float ang = pos * freq;
    const float c = __cosf(ang), sn = __sinf(ang);
    const float* kr = kRaw + ((size_t)(b * KSEL + t)) * (KVH_ * HD_) + kh * HD_;
    float x  = kr[d];
    float xo = (d < 64) ? -kr[d + 64] : kr[d - 64];
    kBf[(((size_t)(b * KVH_ + kh)) * KSEL + t) * HD_ + d] = f2bf(x * c + xo * sn);
    float vv = vRaw[((size_t)(b * KSEL + t)) * (KVH_ * HD_) + kh * HD_ + d];
    vTBf[(((size_t)(b * KVH_ + kh)) * HD_ + d) * KSEL + t] = f2bf(vv);
}

// ---------------- causal GQA attention via WMMA; block = (b,h,64-query tile) ----------------
// 128 threads = 4 waves, each wave owns 16 queries. Two-pass softmax in LDS.
__global__ __launch_bounds__(128) void attn_kernel(
    const unsigned short* __restrict__ qBf, const unsigned short* __restrict__ kBf,
    const unsigned short* __restrict__ vTBf, unsigned short* __restrict__ oBf)
{
    constexpr int SSTR = 516;   // fp32 score row stride
    constexpr int PSTR = 520;   // bf16 prob row stride (16B aligned)
    extern __shared__ char smem[];
    float* scores = reinterpret_cast<float*>(smem);
    unsigned short* probs = reinterpret_cast<unsigned short*>(smem + 64 * SSTR * sizeof(float));

    const int blk = blockIdx.x;
    const int qt = blk & 7;
    const int h  = (blk >> 3) & 15;
    const int b  = blk >> 7;
    const int kvh = h >> 2;           // rep = H/KVH = 4
    const int q0 = qt * 64;
    const int kLen = q0 + 64;         // causal key span (multiple of 32)
    const int tid = threadIdx.x, lane = tid & 31, w = tid >> 5;
    const int nlo = lane & 15, hi = lane >> 4;

    // ---- scores = (q/sqrt(d)) @ k^T ----
    const unsigned short* qrow =
        qBf + (((size_t)(b * H_ + h)) * KSEL + q0 + w * 16 + nlo) * HD_;
    v16bf qf[4];
    for (int d4 = 0; d4 < 4; ++d4) qf[d4] = frag_ld_a(qrow + d4 * 32, lane);
    for (int j0 = 0; j0 < kLen; j0 += 16) {
        const unsigned short* krow =
            kBf + (((size_t)(b * KVH_ + kvh)) * KSEL + j0 + nlo) * HD_;
        v8f sacc = {};
        for (int d4 = 0; d4 < 4; ++d4) {
            v16bf kf = frag_ld_b(krow + d4 * 32, lane);
            sacc = wmma_bf16(qf[d4], kf, sacc);
        }
        for (int r = 0; r < 8; ++r) {
            int qi = q0 + w * 16 + r + 8 * hi;
            int kj = j0 + nlo;
            float val = (kj <= qi) ? sacc[r] : -1e30f;
            scores[(w * 16 + r + 8 * hi) * SSTR + kj] = val;
        }
    }
    __syncthreads();
    // ---- softmax per row ----
    if (tid < 64) {
        float* srow = scores + tid * SSTR;
        float m = -1e30f;
        for (int j = 0; j < kLen; ++j) m = fmaxf(m, srow[j]);
        float sum = 0.f;
        for (int j = 0; j < kLen; ++j) { float e = __expf(srow[j] - m); sum += e; srow[j] = e; }
        float inv = 1.f / sum;
        unsigned short* prow = probs + tid * PSTR;
        for (int j = 0; j < kLen; ++j) prow[j] = f2bf(srow[j] * inv);
    }
    __syncthreads();
    // ---- o = probs @ v^T ----
    v8f oacc[8] = {};
    const unsigned short* prow = probs + (w * 16 + nlo) * PSTR;
    for (int j0 = 0; j0 < kLen; j0 += 32) {
        v16bf pf = frag_ld_a(prow + j0, lane);
        for (int nt = 0; nt < 8; ++nt) {
            const unsigned short* vrow =
                vTBf + (((size_t)(b * KVH_ + kvh)) * HD_ + nt * 16 + nlo) * KSEL;
            v16bf vf = frag_ld_b(vrow + j0, lane);
            oacc[nt] = wmma_bf16(pf, vf, oacc[nt]);
        }
    }
    for (int nt = 0; nt < 8; ++nt)
        for (int r = 0; r < 8; ++r) {
            int tok = q0 + w * 16 + r + 8 * hi;
            int col = h * HD_ + nt * 16 + nlo;
            oBf[((size_t)b * KSEL + tok) * D_ + col] = f2bf(oacc[nt][r]);
        }
}

// ---------------- silu(g)*u -> bf16 ----------------
__global__ __launch_bounds__(256) void silu_fuse_kernel(
    const float* __restrict__ g, const float* __restrict__ u,
    unsigned short* __restrict__ gu, int n)
{
    int i = blockIdx.x * blockDim.x + threadIdx.x;
    int stride = gridDim.x * blockDim.x;
    for (; i < n; i += stride) {
        float gv = g[i];
        float sv = gv / (1.f + __expf(-gv));
        gu[i] = f2bf(sv * u[i]);
    }
}

// ---------------- full passthrough copy of hidden_states ----------------
__global__ __launch_bounds__(256) void copy_kernel(
    const float4* __restrict__ in, float4* __restrict__ out, int n4)
{
    int i = blockIdx.x * blockDim.x + threadIdx.x;
    int stride = gridDim.x * blockDim.x;
    for (; i < n4; i += stride) out[i] = in[i];
}

// ---------------- final: out[b, sel] = (x2 + mlp) * rw_sel ----------------
__global__ __launch_bounds__(256) void scatter_kernel(
    const float* __restrict__ x2, const float* __restrict__ mlpOut,
    const int* __restrict__ selIdx, const float* __restrict__ rwSel, float* __restrict__ out)
{
    const int tk = blockIdx.x;
    const int b = tk >> 9;
    const int si = selIdx[tk];
    const float wgt = rwSel[tk];
    float* orow = out + ((size_t)b * S_ + si) * D_;
    const float* a = x2 + (size_t)tk * D_;
    const float* m = mlpOut + (size_t)tk * D_;
    for (int c = threadIdx.x; c < D_; c += 256)
        orow[c] = (a[c] + m[c]) * wgt;
}

// ============================ host side ============================
extern "C" void kernel_launch(void* const* d_in, const int* in_sizes, int n_in,
                              void* d_out, int out_size, void* d_ws, size_t ws_size,
                              hipStream_t stream) {
    (void)in_sizes; (void)n_in; (void)out_size; (void)ws_size;
    const float* hs  = (const float*)d_in[0];
    // d_in[1] = position_ids: identity arange -> position == selected index, not read.
    const float* wr  = (const float*)d_in[2];
    const float* br  = (const float*)d_in[3];
    const float* ln1 = (const float*)d_in[4];
    const float* wq  = (const float*)d_in[5];
    const float* bq  = (const float*)d_in[6];
    const float* wk  = (const float*)d_in[7];
    const float* bk  = (const float*)d_in[8];
    const float* wv  = (const float*)d_in[9];
    const float* bv  = (const float*)d_in[10];
    const float* wo  = (const float*)d_in[11];
    const float* ln2 = (const float*)d_in[12];
    const float* wg  = (const float*)d_in[13];
    const float* wu  = (const float*)d_in[14];
    const float* wd  = (const float*)d_in[15];
    float* out = (float*)d_out;

    // ---- workspace bump allocator ----
    size_t off = 0;
    auto alloc = [&](size_t bytes) -> void* {
        off = (off + 255) & ~((size_t)255);
        void* p = (char*)d_ws + off;
        off += bytes;
        return p;
    };
    typedef unsigned short us;
    int*   selIdx = (int*)  alloc((size_t)MTOT * 4);
    float* rwSel  = (float*)alloc((size_t)MTOT * 4);
    float* xSel   = (float*)alloc((size_t)MTOT * D_ * 4);
    us*    hBf    = (us*)   alloc((size_t)MTOT * D_ * 2);
    float* qRaw   = (float*)alloc((size_t)MTOT * D_ * 4);           // reused as attnProj
    float* kRaw   = (float*)alloc((size_t)MTOT * KVH_ * HD_ * 4);
    float* vRaw   = (float*)alloc((size_t)MTOT * KVH_ * HD_ * 4);
    us*    qBf    = (us*)   alloc((size_t)B_ * H_ * KSEL * HD_ * 2);
    us*    kBf    = (us*)   alloc((size_t)B_ * KVH_ * KSEL * HD_ * 2);
    us*    vTBf   = (us*)   alloc((size_t)B_ * KVH_ * KSEL * HD_ * 2);
    us*    oBf    = (us*)   alloc((size_t)MTOT * D_ * 2);
    float* x2     = (float*)alloc((size_t)MTOT * D_ * 4);
    us*    h2Bf   = (us*)   alloc((size_t)MTOT * D_ * 2);
    float* gBuf   = (float*)alloc((size_t)MTOT * FF_ * 4);          // reused as mlpOut
    float* uBuf   = (float*)alloc((size_t)MTOT * FF_ * 4);
    us*    guBf   = (us*)   alloc((size_t)MTOT * FF_ * 2);
    us*    wqT    = (us*)   alloc((size_t)D_ * D_ * 2);
    us*    wkT    = (us*)   alloc((size_t)(KVH_ * HD_) * D_ * 2);
    us*    wvT    = (us*)   alloc((size_t)(KVH_ * HD_) * D_ * 2);
    us*    woT    = (us*)   alloc((size_t)D_ * D_ * 2);
    us*    wgT    = (us*)   alloc((size_t)FF_ * D_ * 2);
    us*    wuT    = (us*)   alloc((size_t)FF_ * D_ * 2);
    us*    wdT    = (us*)   alloc((size_t)D_ * FF_ * 2);
    float* attnProj = qRaw;
    float* mlpOut   = gBuf;

    dim3 tb(32, 8);
    // weight convert+transpose: (K x N fp32) -> (N x K bf16)
    w_transpose_bf16<<<dim3(D_ / 32, D_ / 32), tb, 0, stream>>>(wq, wqT, D_, D_);
    w_transpose_bf16<<<dim3((KVH_ * HD_) / 32, D_ / 32), tb, 0, stream>>>(wk, wkT, D_, KVH_ * HD_);
    w_transpose_bf16<<<dim3((KVH_ * HD_) / 32, D_ / 32), tb, 0, stream>>>(wv, wvT, D_, KVH_ * HD_);
    w_transpose_bf16<<<dim3(D_ / 32, D_ / 32), tb, 0, stream>>>(wo, woT, D_, D_);
    w_transpose_bf16<<<dim3(FF_ / 32, D_ / 32), tb, 0, stream>>>(wg, wgT, D_, FF_);
    w_transpose_bf16<<<dim3(FF_ / 32, D_ / 32), tb, 0, stream>>>(wu, wuT, D_, FF_);
    w_transpose_bf16<<<dim3(D_ / 32, FF_ / 32), tb, 0, stream>>>(wd, wdT, FF_, D_);

    router_topk_kernel<<<B_, 512, 0, stream>>>(hs, wr, br, selIdx, rwSel);
    gather_norm_kernel<<<MTOT, 256, 0, stream>>>(hs, selIdx, ln1, xSel, hBf);

    // QKV projections
    gemm_bf16_kernel<<<dim3(D_ / 128, MTOT / 128), 256, 0, stream>>>(hBf, wqT, bq, qRaw, MTOT, D_, D_);
    gemm_bf16_kernel<<<dim3((KVH_ * HD_) / 128, MTOT / 128), 256, 0, stream>>>(hBf, wkT, bk, kRaw, MTOT, KVH_ * HD_, D_);
    gemm_bf16_kernel<<<dim3((KVH_ * HD_) / 128, MTOT / 128), 256, 0, stream>>>(hBf, wvT, bv, vRaw, MTOT, KVH_ * HD_, D_);

    rope_q_kernel<<<B_ * KSEL * H_, 128, 0, stream>>>(qRaw, selIdx, qBf);
    rope_kv_kernel<<<B_ * KSEL * KVH_, 128, 0, stream>>>(kRaw, vRaw, selIdx, kBf, vTBf);

    size_t attnSmem = (size_t)64 * 516 * sizeof(float) + (size_t)64 * 520 * sizeof(unsigned short);
    attn_kernel<<<B_ * H_ * (KSEL / 64), 128, attnSmem, stream>>>(qBf, kBf, vTBf, oBf);

    // O projection + residual + norm2
    gemm_bf16_kernel<<<dim3(D_ / 128, MTOT / 128), 256, 0, stream>>>(oBf, woT, nullptr, attnProj, MTOT, D_, D_);
    resid_norm2_kernel<<<MTOT, 256, 0, stream>>>(xSel, attnProj, ln2, x2, h2Bf);

    // MLP
    gemm_bf16_kernel<<<dim3(FF_ / 128, MTOT / 128), 256, 0, stream>>>(h2Bf, wgT, nullptr, gBuf, MTOT, FF_, D_);
    gemm_bf16_kernel<<<dim3(FF_ / 128, MTOT / 128), 256, 0, stream>>>(h2Bf, wuT, nullptr, uBuf, MTOT, FF_, D_);
    silu_fuse_kernel<<<4096, 256, 0, stream>>>(gBuf, uBuf, guBf, MTOT * FF_);
    gemm_bf16_kernel<<<dim3(D_ / 128, MTOT / 128), 256, 0, stream>>>(guBf, wdT, nullptr, mlpOut, MTOT, D_, FF_);

    // assemble output
    int n4 = (B_ * S_ * D_) / 4;
    copy_kernel<<<8192, 256, 0, stream>>>((const float4*)hs, (float4*)out, n4);
    scatter_kernel<<<MTOT, 256, 0, stream>>>(x2, mlpOut, selIdx, rwSel, out);
}
